// ScaledDotProductAttention_26577257628171
// MI455X (gfx1250) — matrix operand order repaired and checked by reading, not compiled, not run on hardware
//
#include <hip/hip_runtime.h>

// CDNA5 / gfx1250 scaled-dot-product attention.
// B=8, S=2048, E=1024 (single head over full E), f32 in/out.
// Pass 1: convert Q(K,V) to f16 in workspace (V transposed to [E][S]).
// Pass 2: WMMA f16 attention, 32 query rows/block so every K/V B-fragment
//         feeds two accumulator tiles (2x operand reuse vs 16-row blocks).

typedef __attribute__((ext_vector_type(16))) _Float16 v16h;
typedef __attribute__((ext_vector_type(8)))  float    v8f;

#define BATCH   8
#define SEQ     2048
#define EMB     1024
#define QROWS   32
#define NWAVES  8
#define THREADS (NWAVES * 32)

union Frag16 {
    v16h v;
    uint4 u[2];
    _Float16 h[16];
};

// ---------------------------------------------------------------------------
// Pass 1: qkv f32 -> wQ (scaled f16, [B][S][E]), wK (f16, [B][S][E]),
//         wVt (f16, transposed [B][E][S]).  One 32x32 tile per block.
// ---------------------------------------------------------------------------
__global__ __launch_bounds__(256)
void cvt_pack_kernel(const float* __restrict__ qkv,
                     _Float16* __restrict__ wQ,
                     _Float16* __restrict__ wK,
                     _Float16* __restrict__ wVt)
{
    __shared__ _Float16 sT[32][33];            // padded transpose tile

    const int et = blockIdx.x & (EMB / 32 - 1);            // 32 e-tiles
    const int st = (blockIdx.x >> 5) & (SEQ / 32 - 1);     // 64 s-tiles
    const int b  = blockIdx.x >> 11;
    const int e0 = et * 32;
    const int s0 = st * 32;

    const int tx = threadIdx.x & 31;
    const int ty = threadIdx.x >> 5;           // 8 rows, strided by 8

    const float scale = 0.03125f;              // 1/sqrt(1024)

    #pragma unroll
    for (int r = ty; r < 32; r += 8) {
        const size_t srow = (size_t)(b * SEQ + s0 + r) * (3 * EMB);
        const float qv = qkv[srow + e0 + tx];
        const float kv = qkv[srow + EMB + e0 + tx];
        const float vv = qkv[srow + 2 * EMB + e0 + tx];
        wQ[(size_t)(b * SEQ + s0 + r) * EMB + e0 + tx] = (_Float16)(qv * scale);
        wK[(size_t)(b * SEQ + s0 + r) * EMB + e0 + tx] = (_Float16)kv;
        sT[r][tx] = (_Float16)vv;
    }
    __syncthreads();
    #pragma unroll
    for (int r = ty; r < 32; r += 8) {
        wVt[(size_t)(b * EMB + e0 + r) * SEQ + s0 + tx] = sT[tx][r];
    }
}

// ---------------------------------------------------------------------------
// Pass 2: attention. 512 blocks x 256 threads; block = 32 query rows.
// ---------------------------------------------------------------------------
__global__ __launch_bounds__(THREADS)
void sdpa_wmma_kernel(const _Float16* __restrict__ wQ,
                      const _Float16* __restrict__ wK,
                      const _Float16* __restrict__ wVt,
                      float* __restrict__ out)
{
    __shared__ _Float16 sS[QROWS * SEQ];       // 128 KB: scores -> probs
    __shared__ float    sRed[QROWS * 8];
    __shared__ float    sRowSum[QROWS];

    const int tid  = threadIdx.x;
    const int lane = tid & 31;
    const int wave = tid >> 5;
    const int b    = blockIdx.x >> 6;          // SEQ/QROWS = 64 q-tiles/batch
    const int q0   = (blockIdx.x & 63) * QROWS;

    const int hi    = lane >> 4;
    const int lo16  = lane & 15;
    const int klo8  = hi * 8;                  // A-fragment half-wave stagger
    const int klo16 = hi * 16;                 // unused (B fragments use klo8 as well)

    // Per-lane A row pointers for the two 16-row halves of the Q tile.
    const _Float16* qrow0 = wQ + (size_t)(b * SEQ + q0 + lo16) * EMB;
    const _Float16* qrow1 = qrow0 + (size_t)16 * EMB;

    // ---- Phase 1: S[32][2048] = Qs . K^T -----------------------------------
    // Each wave: 4 key-tiles (A reused x4) x 2 row-halves (B reused x2).
    for (int kg = 0; kg < 4; ++kg) {
        const int base = kg * 512 + wave * 64;
        v8f acc[2][4];
        #pragma unroll
        for (int h = 0; h < 2; ++h)
            #pragma unroll
            for (int c = 0; c < 4; ++c) { v8f z = {}; acc[h][c] = z; }

        const _Float16* krow0 = wK + (size_t)(b * SEQ + base + lo16) * EMB;
        __builtin_prefetch(krow0, 0, 0);       // global_prefetch_b8

        #pragma unroll 2
        for (int t = 0; t < 32; ++t) {
            const int ebase = t * 32;
            Frag16 a0, a1;                     // A: Q 16x32, rows m = lo16 (+16)
            a0.u[0] = *(const uint4*)(qrow0 + ebase + klo8);
            a0.u[1] = *(const uint4*)(qrow0 + ebase + 16 + klo8);
            a1.u[0] = *(const uint4*)(qrow1 + ebase + klo8);
            a1.u[1] = *(const uint4*)(qrow1 + ebase + 16 + klo8);
            #pragma unroll
            for (int c = 0; c < 4; ++c) {
                // B: K^T 32x16, col n = lo16; E halves in table order (8h+s, 16+8h+s)
                const _Float16* kp = krow0 + (size_t)(c * 16) * EMB + ebase + klo8;
                Frag16 kf;
                kf.u[0] = ((const uint4*)kp)[0];
                kf.u[1] = ((const uint4*)kp)[2];
                acc[0][c] = __builtin_amdgcn_wmma_f32_16x16x32_f16(
                    false, a0.v, false, kf.v, (short)0, acc[0][c], false, false);
                acc[1][c] = __builtin_amdgcn_wmma_f32_16x16x32_f16(
                    false, a1.v, false, kf.v, (short)0, acc[1][c], false, false);
            }
        }
        #pragma unroll
        for (int h = 0; h < 2; ++h)
            #pragma unroll
            for (int c = 0; c < 4; ++c)
                #pragma unroll
                for (int r = 0; r < 8; ++r) {
                    const int m = h * 16 + r + 8 * hi;   // C layout
                    sS[m * SEQ + base + c * 16 + lo16] = (_Float16)acc[h][c][r];
                }
    }
    __syncthreads();

    // ---- Phase 2: softmax over rows of sS (in place, f16) -------------------
    {
        const int row = tid >> 3;              // 32 rows
        const int seg = tid & 7;               // 8 segments x 256 cols
        const int c0  = seg * 256;
        float lmax = -1e30f;
        for (int c = 0; c < 256; ++c)
            lmax = fmaxf(lmax, (float)sS[row * SEQ + c0 + c]);
        sRed[row * 8 + seg] = lmax;
        __syncthreads();
        float rmax = -1e30f;
        #pragma unroll
        for (int j = 0; j < 8; ++j)
            rmax = fmaxf(rmax, sRed[row * 8 + j]);
        __syncthreads();
        float lsum = 0.f;
        for (int c = 0; c < 256; ++c) {
            const int idx = row * SEQ + c0 + c;
            const float p = __expf((float)sS[idx] - rmax);
            sS[idx] = (_Float16)p;
            lsum += p;
        }
        sRed[row * 8 + seg] = lsum;
        __syncthreads();
        if (seg == 0) {
            float s = 0.f;
            #pragma unroll
            for (int j = 0; j < 8; ++j) s += sRed[row * 8 + j];
            sRowSum[row] = s;
        }
        __syncthreads();
    }

    // ---- Phase 3: O slice = P . Vt  (per-wave 128-wide E slice, 32 rows) ----
    const int ew = wave * 128;
    v8f o[2][8];
    #pragma unroll
    for (int h = 0; h < 2; ++h)
        #pragma unroll
        for (int t = 0; t < 8; ++t) { v8f z = {}; o[h][t] = z; }

    const _Float16* vtbase = wVt + (size_t)(b * EMB + ew) * SEQ;

    for (int kt = 0; kt < 64; ++kt) {
        const int kb = kt * 32;
        Frag16 p0, p1;                         // A: P 16x32, halves over keys
        p0.u[0] = *(const uint4*)&sS[(lo16)      * SEQ + kb + klo8];
        p0.u[1] = *(const uint4*)&sS[(lo16)      * SEQ + kb + 16 + klo8];
        p1.u[0] = *(const uint4*)&sS[(16 + lo16) * SEQ + kb + klo8];
        p1.u[1] = *(const uint4*)&sS[(16 + lo16) * SEQ + kb + 16 + klo8];
        #pragma unroll
        for (int t = 0; t < 8; ++t) {
            // B: V 32x16, col n = e = ew+16t+lo16; keys in table order
            const _Float16* vp = vtbase + (size_t)(16 * t + lo16) * SEQ + kb + klo8;
            Frag16 vf;
            vf.u[0] = ((const uint4*)vp)[0];
            vf.u[1] = ((const uint4*)vp)[2];
            o[0][t] = __builtin_amdgcn_wmma_f32_16x16x32_f16(
                false, p0.v, false, vf.v, (short)0, o[0][t], false, false);
            o[1][t] = __builtin_amdgcn_wmma_f32_16x16x32_f16(
                false, p1.v, false, vf.v, (short)0, o[1][t], false, false);
        }
    }

    // ---- Write-out with deferred softmax normalization ----------------------
    float* obase = out + (size_t)(b * SEQ + q0) * EMB;
    #pragma unroll
    for (int h = 0; h < 2; ++h)
        #pragma unroll
        for (int r = 0; r < 8; ++r) {
            const int m = h * 16 + r + 8 * hi;
            const float inv = 1.0f / sRowSum[m];
            #pragma unroll
            for (int t = 0; t < 8; ++t) {
                const int e = ew + 16 * t + lo16;
                obase[(size_t)m * EMB + e] = o[h][t][r] * inv;
            }
        }
}

extern "C" void kernel_launch(void* const* d_in, const int* in_sizes, int n_in,
                              void* d_out, int out_size, void* d_ws, size_t ws_size,
                              hipStream_t stream)
{
    const float* qkv = (const float*)d_in[0];
    float* out = (float*)d_out;

    _Float16* wQ  = (_Float16*)d_ws;
    _Float16* wK  = wQ + (size_t)BATCH * SEQ * EMB;
    _Float16* wVt = wK + (size_t)BATCH * SEQ * EMB;   // total 96 MB of d_ws

    const dim3 cgrid(BATCH * (SEQ / 32) * (EMB / 32));  // 16384 blocks
    cvt_pack_kernel<<<cgrid, 256, 0, stream>>>(qkv, wQ, wK, wVt);

    const dim3 grid(BATCH * (SEQ / QROWS));             // 512 blocks
    sdpa_wmma_kernel<<<grid, THREADS, 0, stream>>>(wQ, wK, wVt, out);
}
